// OpenMythos_32177894981784
// MI455X (gfx1250) — matrix operand order repaired and checked by reading, not compiled, not run on hardware
//
#include <hip/hip_runtime.h>
#include <hip/hip_bf16.h>

// ---------------- model constants ----------------
#define V_SZ    32000
#define DIM     1024
#define NH      16
#define QK_NOPE 64
#define QK_ROPE 32
#define VHD     64
#define KV_RANK 256
#define Q_RANK  512
#define NE      8
#define ED      512
#define LORA_R  16
#define LOOP_DIM 64
#define T_LOOP  4
#define THETA   500000.0f
#define ACT_TH  0.99f
#define BSZ     4
#define SEQ     512
#define NTOK    (BSZ*SEQ)
#define QHD     (QK_NOPE+QK_ROPE)   /* 96  */
#define KVP_D   (QK_NOPE+VHD)       /* 128 */
#define KVA_D   (KV_RANK+QK_ROPE)   /* 288 */

typedef __attribute__((ext_vector_type(16))) __bf16 bf16x16;
typedef __attribute__((ext_vector_type(8)))  float  floatx8;

// =====================================================================
// Batched WMMA GEMM:  C[z] = alpha * A[z] (MxK, f32) * B[z] (KxN)
// B may be bf16 or f32, row-major (KxN, ldb) or transposed (NxK, ldb).
// One 256-thread block computes a 32x64 C tile; 8 waves -> 8 16x16 WMMA
// tiles; K marched in chunks of 32 through LDS with f32->bf16 convert.
// A/B fragments use contiguous k = 16*(lane/16) + s in both operands (wave32).
// =====================================================================
template<int B_BF16, int B_TRANS>
__global__ __launch_bounds__(256) void mythos_wmma_gemm(
    const float* __restrict__ A, const void* __restrict__ Bp,
    float* __restrict__ C, int M, int N, int K,
    long sA, long sB, long sC, int lda, int ldb, int ldc, float alpha)
{
  __shared__ __bf16 shA[32][32];
  __shared__ __bf16 shB[32][64];
  const int z = blockIdx.z;
  const float*  Ab = A + (long)z * sA;
  const float*  Bf = (const float*)Bp  + (long)z * sB;
  const __bf16* Bb = (const __bf16*)Bp + (long)z * sB;
  float* Cb = C + (long)z * sC;
  const int tid  = threadIdx.x;
  const int lane = tid & 31;
  const int wave = tid >> 5;
  const int m0 = blockIdx.y * 32;
  const int n0 = blockIdx.x * 64;
  const int wm = (wave >> 2) * 16;   // 0 / 16
  const int wn = (wave & 3) * 16;    // 0 / 16 / 32 / 48
  floatx8 acc = {0.f,0.f,0.f,0.f,0.f,0.f,0.f,0.f};

  for (int k0 = 0; k0 < K; k0 += 32) {
    // stage A tile (32x32) with f32->bf16 convert
    for (int i = tid; i < 32*32; i += 256) {
      int r = i >> 5, c = i & 31;
      int gr = m0 + r, gc = k0 + c;
      float v = (gr < M && gc < K) ? Ab[(long)gr * lda + gc] : 0.f;
      shA[r][c] = (__bf16)v;
    }
    // stage B tile (32x64)
    for (int i = tid; i < 32*64; i += 256) {
      int r = i >> 6, c = i & 63;            // r = k, c = n
      int gk = k0 + r, gn = n0 + c;
      float v = 0.f;
      if (gk < K && gn < N) {
        long off = B_TRANS ? ((long)gn * ldb + gk) : ((long)gk * ldb + gn);
        v = B_BF16 ? (float)Bb[off] : Bf[off];
      }
      shB[r][c] = (__bf16)v;
    }
    // prefetch next-K A rows (lowers to global_prefetch_b8 on gfx1250)
    if (k0 + 32 < K && wave == 0) {
      int gr = m0 + lane;
      if (gr < M) __builtin_prefetch(&Ab[(long)gr * lda + k0 + 32], 0, 1);
    }
    __syncthreads();

    bf16x16 fa, fb;
    {
      // A 16x32 bf16 fragment: lanes 0-15 row M=lane, K 0..15;
      // lanes 16-31 same rows, K 16..31
      int row = wm + (lane & 15);
      int kb  = (lane >> 4) * 16;
#pragma unroll
      for (int j = 0; j < 8; ++j) {
        fa[j]     = shA[row][kb + j];
        fa[8 + j] = shA[row][kb + 8 + j];
      }
      // B 32x16 bf16 fragment: lanes 0-15 col N=lane K=0..15,
      // lanes 16-31 K=16..31
      int col = wn + (lane & 15);
      int kb2 = (lane >> 4) * 16;
#pragma unroll
      for (int j = 0; j < 16; ++j) fb[j] = shB[kb2 + j][col];
    }
    acc = __builtin_amdgcn_wmma_f32_16x16x32_bf16(
        false, fa, false, fb, (short)0, acc, false, false);
    __syncthreads();
  }
  // C/D f32 16x16: VGPR v, lanes 0-15 -> M = v, lanes 16-31 -> M = 8+v
  {
    int col = n0 + wn + (lane & 15);
    int rb  = (lane >> 4) * 8;
#pragma unroll
    for (int v = 0; v < 8; ++v) {
      int row = m0 + wm + rb + v;
      if (row < M && col < N) Cb[(long)row * ldc + col] = acc[v] * alpha;
    }
  }
}

// ---------------- elementwise / reduction kernels ----------------
__global__ void k_conv_bf16(const float* __restrict__ s, __bf16* __restrict__ d, long n) {
  long i = (long)blockIdx.x * 256 + threadIdx.x;
  if (i < n) d[i] = (__bf16)s[i];
}

__global__ void k_embed(const int* __restrict__ tok, const float* __restrict__ emb,
                        float* __restrict__ h) {
  int n = blockIdx.x;
  long src = (long)tok[n] * DIM;
  for (int i = threadIdx.x; i < DIM; i += 256) h[(long)n * DIM + i] = emb[src + i];
}

__global__ void k_rmsnorm(const float* __restrict__ x, const float* __restrict__ w,
                          float* __restrict__ y, int D, int xs, int ys) {
  __shared__ float red[256];
  long row = blockIdx.x;
  const float* xr = x + row * (long)xs;
  float s = 0.f;
  for (int i = threadIdx.x; i < D; i += 256) { float v = xr[i]; s += v * v; }
  red[threadIdx.x] = s; __syncthreads();
  for (int o = 128; o > 0; o >>= 1) { if (threadIdx.x < o) red[threadIdx.x] += red[threadIdx.x + o]; __syncthreads(); }
  float r = rsqrtf(red[0] / (float)D + 1e-6f);
  for (int i = threadIdx.x; i < D; i += 256) y[row * (long)ys + i] = xr[i] * r * w[i];
}

__global__ void k_gemv(const float* __restrict__ X, const float* __restrict__ W,
                       const float* __restrict__ bias, float* __restrict__ Y,
                       int K, int NC) {
  __shared__ float red[256];
  const float* x = X + (long)blockIdx.x * K;
  for (int c = 0; c < NC; ++c) {
    float s = 0.f;
    for (int k = threadIdx.x; k < K; k += 256) s += x[k] * W[(long)k * NC + c];
    red[threadIdx.x] = s; __syncthreads();
    for (int o = 128; o > 0; o >>= 1) { if (threadIdx.x < o) red[threadIdx.x] += red[threadIdx.x + o]; __syncthreads(); }
    if (threadIdx.x == 0) Y[(long)blockIdx.x * NC + c] = red[0] + (bias ? bias[c] : 0.f);
    __syncthreads();
  }
}

__global__ void k_rope_q(float* __restrict__ q) {
  long idx = (long)blockIdx.x * 256 + threadIdx.x;
  if (idx >= (long)NTOK * NH * (QK_ROPE / 2)) return;
  int i = idx % (QK_ROPE / 2);
  int h = (idx / (QK_ROPE / 2)) % NH;
  long n = idx / ((long)NH * (QK_ROPE / 2));
  int pos = (int)(n % SEQ);
  float f = powf(THETA, -(2.0f * i) / (float)QK_ROPE);
  float ang = pos * f, c = cosf(ang), s = sinf(ang);
  long base = (n * NH + h) * (long)QHD + QK_NOPE + 2 * i;
  float x1 = q[base], x2 = q[base + 1];
  q[base] = x1 * c - x2 * s;
  q[base + 1] = x1 * s + x2 * c;
}

__global__ void k_rope_kpe(float* __restrict__ kv) {
  long idx = (long)blockIdx.x * 256 + threadIdx.x;
  if (idx >= (long)NTOK * (QK_ROPE / 2)) return;
  int i = idx % (QK_ROPE / 2);
  long n = idx / (QK_ROPE / 2);
  int pos = (int)(n % SEQ);
  float f = powf(THETA, -(2.0f * i) / (float)QK_ROPE);
  float ang = pos * f, c = cosf(ang), s = sinf(ang);
  long base = n * (long)KVA_D + KV_RANK + 2 * i;
  float x1 = kv[base], x2 = kv[base + 1];
  kv[base] = x1 * c - x2 * s;
  kv[base + 1] = x1 * s + x2 * c;
}

__global__ void k_pack_q(const float* __restrict__ q, float* __restrict__ qp) {
  long idx = (long)blockIdx.x * 256 + threadIdx.x;
  if (idx >= (long)NTOK * NH * QHD) return;
  int d = idx % QHD; int h = (idx / QHD) % NH; long n = idx / ((long)NH * QHD);
  long b = n / SEQ, t = n % SEQ;
  qp[(((b * NH + h) * SEQ) + t) * QHD + d] = q[(n * NH + h) * (long)QHD + d];
}

__global__ void k_build_k(const float* __restrict__ kvp, const float* __restrict__ kv,
                          float* __restrict__ kp) {
  long idx = (long)blockIdx.x * 256 + threadIdx.x;
  if (idx >= (long)NTOK * NH * QHD) return;
  int d = idx % QHD; int h = (idx / QHD) % NH; long n = idx / ((long)NH * QHD);
  long b = n / SEQ, t = n % SEQ;
  float v = (d < QK_NOPE) ? kvp[n * (long)(NH * KVP_D) + h * KVP_D + d]
                          : kv[n * (long)KVA_D + KV_RANK + (d - QK_NOPE)];
  kp[(((b * NH + h) * SEQ) + t) * QHD + d] = v;
}

__global__ void k_pack_v(const float* __restrict__ kvp, float* __restrict__ vp) {
  long idx = (long)blockIdx.x * 256 + threadIdx.x;
  if (idx >= (long)NTOK * NH * VHD) return;
  int d = idx % VHD; int h = (idx / VHD) % NH; long n = idx / ((long)NH * VHD);
  long b = n / SEQ, t = n % SEQ;
  vp[(((b * NH + h) * SEQ) + t) * VHD + d] = kvp[n * (long)(NH * KVP_D) + h * KVP_D + QK_NOPE + d];
}

__global__ void k_softmax_causal(float* __restrict__ sc) {
  __shared__ float red[256];
  long z = blockIdx.y; int q = blockIdx.x;
  float* row = sc + (z * SEQ + q) * (long)SEQ;
  int nall = q + 1;
  float m = -1e30f;
  for (int i = threadIdx.x; i < nall; i += 256) m = fmaxf(m, row[i]);
  red[threadIdx.x] = m; __syncthreads();
  for (int o = 128; o > 0; o >>= 1) { if (threadIdx.x < o) red[threadIdx.x] = fmaxf(red[threadIdx.x], red[threadIdx.x + o]); __syncthreads(); }
  float M = red[0]; __syncthreads();
  float s = 0.f;
  for (int i = threadIdx.x; i < nall; i += 256) { float e = __expf(row[i] - M); row[i] = e; s += e; }
  red[threadIdx.x] = s; __syncthreads();
  for (int o = 128; o > 0; o >>= 1) { if (threadIdx.x < o) red[threadIdx.x] += red[threadIdx.x + o]; __syncthreads(); }
  float inv = 1.f / red[0];
  for (int i = threadIdx.x; i < SEQ; i += 256) row[i] = (i < nall) ? row[i] * inv : 0.f;
}

__global__ void k_unpack_attn(const float* __restrict__ op, float* __restrict__ ax) {
  long idx = (long)blockIdx.x * 256 + threadIdx.x;
  if (idx >= (long)NTOK * NH * VHD) return;
  int d = idx % VHD; int h = (idx / VHD) % NH; long n = idx / ((long)NH * VHD);
  long b = n / SEQ, t = n % SEQ;
  ax[n * (long)DIM + h * VHD + d] = op[(((b * NH + h) * SEQ) + t) * VHD + d];
}

__global__ void k_topk(const float* __restrict__ rl, float* __restrict__ wf) {
  int n = blockIdx.x * 256 + threadIdx.x;
  if (n >= NTOK) return;
  float p[NE]; float m = -1e30f;
  for (int e = 0; e < NE; ++e) { p[e] = rl[(long)n * NE + e]; m = fmaxf(m, p[e]); }
  float s = 0.f;
  for (int e = 0; e < NE; ++e) { p[e] = __expf(p[e] - m); s += p[e]; }
  for (int e = 0; e < NE; ++e) p[e] /= s;
  int i0 = 0; for (int e = 1; e < NE; ++e) if (p[e] > p[i0]) i0 = e;
  int i1 = (i0 == 0) ? 1 : 0;
  for (int e = 0; e < NE; ++e) if (e != i0 && p[e] > p[i1]) i1 = e;
  float ws = p[i0] + p[i1];
  for (int e = 0; e < NE; ++e) wf[(long)n * NE + e] = 0.f;
  wf[(long)n * NE + i0] = p[i0] / ws;
  wf[(long)n * NE + i1] = p[i1] / ws;
}

__global__ void k_silu_mul(float* __restrict__ g, const float* __restrict__ u, long n) {
  long i = (long)blockIdx.x * 256 + threadIdx.x;
  if (i >= n) return;
  float x = g[i];
  g[i] = (x / (1.f + __expf(-x))) * u[i];
}

__global__ void k_moe_combine(const float* __restrict__ ym, const float* __restrict__ wf,
                              const float* __restrict__ sh, float* __restrict__ out) {
  long idx = (long)blockIdx.x * 256 + threadIdx.x;
  if (idx >= (long)NTOK * DIM) return;
  long n = idx / DIM; int d = idx % DIM;
  float s = sh[idx];
  for (int e = 0; e < NE; ++e)
    s += wf[n * NE + e] * ym[(n * NE + e) * (long)DIM + d];
  out[idx] = s;
}

__global__ void k_add(float* __restrict__ a, const float* __restrict__ b, long n) {
  long i = (long)blockIdx.x * 256 + threadIdx.x;
  if (i < n) a[i] += b[i];
}
__global__ void k_add_out(float* __restrict__ d, const float* __restrict__ a,
                          const float* __restrict__ b, long n) {
  long i = (long)blockIdx.x * 256 + threadIdx.x;
  if (i < n) d[i] = a[i] + b[i];
}
__global__ void k_copy(float* __restrict__ d, const float* __restrict__ s, long n) {
  long i = (long)blockIdx.x * 256 + threadIdx.x;
  if (i < n) d[i] = s[i];
}

__global__ void k_loop_emb(const float* __restrict__ h, float* __restrict__ ht, int t) {
  long idx = (long)blockIdx.x * 256 + threadIdx.x;
  if (idx >= (long)NTOK * DIM) return;
  int d = idx % DIM;
  float v = h[idx];
  if (d < LOOP_DIM) {
    int i = (d < LOOP_DIM / 2) ? d : d - LOOP_DIM / 2;
    float f = powf(10000.f, -(2.0f * i) / (float)LOOP_DIM);
    float ang = (float)t * f;
    v += (d < LOOP_DIM / 2) ? sinf(ang) : cosf(ang);
  }
  ht[idx] = v;
}

__global__ void k_avec(const float* __restrict__ logA, const float* __restrict__ logdt,
                       float* __restrict__ A) {
  int d = blockIdx.x * 256 + threadIdx.x;
  if (d >= DIM) return;
  float x = logdt[0] + logA[d];
  x = fminf(20.f, fmaxf(-20.f, x));
  A[d] = __expf(-__expf(x));
}

__global__ void k_recurrence(float* __restrict__ h, const float* __restrict__ A,
                             const float* __restrict__ Bv, const float* __restrict__ e,
                             const float* __restrict__ tb) {
  long idx = (long)blockIdx.x * 256 + threadIdx.x;
  if (idx >= (long)NTOK * DIM) return;
  int d = idx % DIM;
  h[idx] = A[d] * h[idx] + Bv[d] * e[idx] + tb[idx];
}

__global__ void k_halt(const float* __restrict__ raw, float* __restrict__ cum,
                       float* __restrict__ wv, int t) {
  int n = blockIdx.x * 256 + threadIdx.x;
  if (n >= NTOK) return;
  float p = 1.f / (1.f + __expf(-raw[n]));
  float c = cum[n];
  bool still = c < ACT_TH;
  float pn = still ? p : 0.f;
  float nc = c + pn;
  bool hn = still && ((nc >= ACT_TH) || (t == T_LOOP - 1));
  wv[n] = hn ? (1.f - c) : pn;
  cum[n] = nc;
}

__global__ void k_axpy_rows(float* __restrict__ out, const float* __restrict__ wv,
                            const float* __restrict__ h) {
  long idx = (long)blockIdx.x * 256 + threadIdx.x;
  if (idx >= (long)NTOK * DIM) return;
  out[idx] += wv[idx / DIM] * h[idx];
}

__global__ void k_scale_cols(float* __restrict__ t16, const float* __restrict__ sc, long n) {
  long i = (long)blockIdx.x * 256 + threadIdx.x;
  if (i < n) t16[i] *= sc[i % LORA_R];
}

// ---------------- host orchestration ----------------
static inline unsigned gblk(long n) { return (unsigned)((n + 255) / 256); }

struct BC {
  __bf16 *wq_a, *wq_b, *wkv_a, *wkv_b, *wo, *wg, *wu, *wd, *sg, *su, *sd;
};

extern "C" void kernel_launch(void* const* d_in, const int* in_sizes, int n_in,
                              void* d_out, int out_size, void* d_ws, size_t ws_size,
                              hipStream_t stream) {
  (void)in_sizes; (void)n_in; (void)out_size; (void)ws_size;
  const int*   tokens = (const int*)d_in[0];
  const float* embed  = (const float*)d_in[1];
  const float* lti_logA = (const float*)d_in[50];
  const float* lti_logdt = (const float*)d_in[51];
  const float* lti_B  = (const float*)d_in[52];
  const float* lora_scale = (const float*)d_in[55];
  const float* halt_w = (const float*)d_in[56];
  const float* halt_b = (const float*)d_in[57];
  const float* norm_f = (const float*)d_in[58];

  char* wsb = (char*)d_ws; size_t cur = 0;
  auto alloc  = [&](size_t bytes) -> void* { size_t o = (cur + 255) & ~(size_t)255; cur = o + bytes; return (void*)(wsb + o); };
  auto allocf = [&](long n) -> float*  { return (float*)alloc(sizeof(float) * n); };
  auto allocb = [&](long n) -> __bf16* { return (__bf16*)alloc(sizeof(__bf16) * n); };
  auto conv = [&](const void* s, __bf16* d, long n) {
    k_conv_bf16<<<gblk(n), 256, 0, stream>>>((const float*)s, d, n);
  };

  // -------- bf16 weight caches --------
  BC cs[3]; const int bases[3] = {2, 18, 34};
  for (int bi = 0; bi < 3; ++bi) {
    int b = bases[bi]; BC& c = cs[bi];
    c.wq_a  = allocb((long)DIM * Q_RANK);        conv(d_in[b + 0], c.wq_a,  (long)DIM * Q_RANK);
    c.wq_b  = allocb((long)Q_RANK * NH * QHD);   conv(d_in[b + 2], c.wq_b,  (long)Q_RANK * NH * QHD);
    c.wkv_a = allocb((long)DIM * KVA_D);         conv(d_in[b + 3], c.wkv_a, (long)DIM * KVA_D);
    c.wkv_b = allocb((long)KV_RANK * NH * KVP_D);conv(d_in[b + 5], c.wkv_b, (long)KV_RANK * NH * KVP_D);
    c.wo    = allocb((long)DIM * DIM);           conv(d_in[b + 6], c.wo,    (long)DIM * DIM);
    c.wg    = allocb((long)NE * DIM * ED);       conv(d_in[b + 8], c.wg,    (long)NE * DIM * ED);
    c.wu    = allocb((long)NE * DIM * ED);       conv(d_in[b + 9], c.wu,    (long)NE * DIM * ED);
    c.wd    = allocb((long)NE * ED * DIM);       conv(d_in[b + 10], c.wd,   (long)NE * ED * DIM);
    c.sg    = allocb((long)DIM * ED);            conv(d_in[b + 11], c.sg,   (long)DIM * ED);
    c.su    = allocb((long)DIM * ED);            conv(d_in[b + 12], c.su,   (long)DIM * ED);
    c.sd    = allocb((long)ED * DIM);            conv(d_in[b + 13], c.sd,   (long)ED * DIM);
  }
  __bf16* embed_bf = allocb((long)V_SZ * DIM);   conv(embed, embed_bf, (long)V_SZ * DIM);
  __bf16* ldown_bf = allocb((long)DIM * LORA_R); conv(d_in[53], ldown_bf, (long)DIM * LORA_R);
  __bf16* lB_bf    = allocb((long)LORA_R * DIM); conv(d_in[54], lB_bf, (long)LORA_R * DIM);

  // -------- activation buffers --------
  float* xn   = allocf((long)NTOK * DIM);
  float* qa   = allocf((long)NTOK * Q_RANK);
  float* qn   = allocf((long)NTOK * Q_RANK);
  float* qbuf = allocf((long)NTOK * NH * QHD);
  float* kvb  = allocf((long)NTOK * KVA_D);
  float* latn = allocf((long)NTOK * KV_RANK);
  float* kvp  = allocf((long)NTOK * NH * KVP_D);
  float* qpk  = allocf((long)NTOK * NH * QHD);
  float* kpk  = allocf((long)NTOK * NH * QHD);
  float* vpk  = allocf((long)NTOK * NH * VHD);
  float* scr  = allocf((long)BSZ * NH * SEQ * SEQ);
  float* opk  = allocf((long)NTOK * NH * VHD);
  float* attx = allocf((long)NTOK * DIM);
  float* att  = allocf((long)NTOK * DIM);
  float* rl   = allocf((long)NTOK * NE);
  float* wfl  = allocf((long)NTOK * NE);
  float* gbuf = allocf((long)NTOK * NE * ED);
  float* ubuf = allocf((long)NTOK * NE * ED);
  float* ym   = allocf((long)NTOK * NE * DIM);
  float* gs   = allocf((long)NTOK * ED);
  float* us   = allocf((long)NTOK * ED);
  float* sot  = allocf((long)NTOK * DIM);
  float* mo   = allocf((long)NTOK * DIM);
  float* h0   = allocf((long)NTOK * DIM);
  float* h1   = allocf((long)NTOK * DIM);
  float* ebuf = allocf((long)NTOK * DIM);
  float* ht   = allocf((long)NTOK * DIM);
  float* tb   = allocf((long)NTOK * DIM);
  float* oac  = allocf((long)NTOK * DIM);
  float* lor  = allocf((long)NTOK * DIM);
  float* Avec = allocf(DIM);
  float* cum  = allocf(NTOK);
  float* wv   = allocf(NTOK);
  float* hw   = allocf(NTOK);
  float* t16  = allocf((long)NTOK * LORA_R);

  // -------- GEMM dispatch helpers --------
  auto gemm = [&](int bbf, int bt, const float* A, const void* Bp, float* C,
                  int M, int N, int K, int lda, int ldb, int ldc,
                  long sA, long sB, long sC, int batches, float alpha) {
    dim3 g((N + 63) / 64, (M + 31) / 32, batches);
    if (bbf) {
      if (bt) mythos_wmma_gemm<1, 1><<<g, 256, 0, stream>>>(A, Bp, C, M, N, K, sA, sB, sC, lda, ldb, ldc, alpha);
      else    mythos_wmma_gemm<1, 0><<<g, 256, 0, stream>>>(A, Bp, C, M, N, K, sA, sB, sC, lda, ldb, ldc, alpha);
    } else {
      if (bt) mythos_wmma_gemm<0, 1><<<g, 256, 0, stream>>>(A, Bp, C, M, N, K, sA, sB, sC, lda, ldb, ldc, alpha);
      else    mythos_wmma_gemm<0, 0><<<g, 256, 0, stream>>>(A, Bp, C, M, N, K, sA, sB, sC, lda, ldb, ldc, alpha);
    }
  };

  const float inv_sqrt_d = 1.0f / sqrtf((float)QHD);
  const long ND = (long)NTOK * DIM;

  // -------- one transformer block: xdst = block(xsrc) --------
  auto run_block = [&](const float* xsrc, float* xdst, int base, const BC& c) {
    const float* q_norm  = (const float*)d_in[base + 1];
    const float* kv_norm = (const float*)d_in[base + 4];
    const float* router  = (const float*)d_in[base + 7];
    const float* norm1   = (const float*)d_in[base + 14];
    const float* norm2   = (const float*)d_in[base + 15];

    // ---- MLA attention ----
    k_rmsnorm<<<NTOK, 256, 0, stream>>>(xsrc, norm1, xn, DIM, DIM, DIM);
    gemm(1, 0, xn, c.wq_a, qa, NTOK, Q_RANK, DIM, DIM, Q_RANK, Q_RANK, 0, 0, 0, 1, 1.f);
    k_rmsnorm<<<NTOK, 256, 0, stream>>>(qa, q_norm, qn, Q_RANK, Q_RANK, Q_RANK);
    gemm(1, 0, qn, c.wq_b, qbuf, NTOK, NH * QHD, Q_RANK, Q_RANK, NH * QHD, NH * QHD, 0, 0, 0, 1, 1.f);
    k_rope_q<<<gblk((long)NTOK * NH * (QK_ROPE / 2)), 256, 0, stream>>>(qbuf);
    gemm(1, 0, xn, c.wkv_a, kvb, NTOK, KVA_D, DIM, DIM, KVA_D, KVA_D, 0, 0, 0, 1, 1.f);
    k_rope_kpe<<<gblk((long)NTOK * (QK_ROPE / 2)), 256, 0, stream>>>(kvb);
    k_rmsnorm<<<NTOK, 256, 0, stream>>>(kvb, kv_norm, latn, KV_RANK, KVA_D, KV_RANK);
    gemm(1, 0, latn, c.wkv_b, kvp, NTOK, NH * KVP_D, KV_RANK, KV_RANK, NH * KVP_D, NH * KVP_D, 0, 0, 0, 1, 1.f);
    k_pack_q<<<gblk((long)NTOK * NH * QHD), 256, 0, stream>>>(qbuf, qpk);
    k_build_k<<<gblk((long)NTOK * NH * QHD), 256, 0, stream>>>(kvp, kvb, kpk);
    k_pack_v<<<gblk((long)NTOK * NH * VHD), 256, 0, stream>>>(kvp, vpk);
    // scores[z] = q[z] (SxD) * k[z]^T, z = b*H+h
    gemm(0, 1, qpk, kpk, scr, SEQ, SEQ, QHD, QHD, QHD, SEQ,
         (long)SEQ * QHD, (long)SEQ * QHD, (long)SEQ * SEQ, BSZ * NH, inv_sqrt_d);
    k_softmax_causal<<<dim3(SEQ, BSZ * NH), 256, 0, stream>>>(scr);
    gemm(0, 0, scr, vpk, opk, SEQ, VHD, SEQ, SEQ, VHD, VHD,
         (long)SEQ * SEQ, (long)SEQ * VHD, (long)SEQ * VHD, BSZ * NH, 1.f);
    k_unpack_attn<<<gblk((long)NTOK * NH * VHD), 256, 0, stream>>>(opk, attx);
    gemm(1, 0, attx, c.wo, att, NTOK, DIM, DIM, DIM, DIM, DIM, 0, 0, 0, 1, 1.f);
    k_add_out<<<gblk(ND), 256, 0, stream>>>(xdst, xsrc, att, ND);

    // ---- MoE FFN ----
    k_rmsnorm<<<NTOK, 256, 0, stream>>>(xdst, norm2, xn, DIM, DIM, DIM);
    k_gemv<<<NTOK, 256, 0, stream>>>(xn, router, nullptr, rl, DIM, NE);
    k_topk<<<gblk(NTOK), 256, 0, stream>>>(rl, wfl);
    gemm(1, 0, xn, c.wg, gbuf, NTOK, ED, DIM, DIM, ED, NE * ED, 0, (long)DIM * ED, ED, NE, 1.f);
    gemm(1, 0, xn, c.wu, ubuf, NTOK, ED, DIM, DIM, ED, NE * ED, 0, (long)DIM * ED, ED, NE, 1.f);
    k_silu_mul<<<gblk((long)NTOK * NE * ED), 256, 0, stream>>>(gbuf, ubuf, (long)NTOK * NE * ED);
    gemm(1, 0, gbuf, c.wd, ym, NTOK, DIM, ED, NE * ED, DIM, NE * DIM, ED, (long)ED * DIM, DIM, NE, 1.f);
    gemm(1, 0, xn, c.sg, gs, NTOK, ED, DIM, DIM, ED, ED, 0, 0, 0, 1, 1.f);
    gemm(1, 0, xn, c.su, us, NTOK, ED, DIM, DIM, ED, ED, 0, 0, 0, 1, 1.f);
    k_silu_mul<<<gblk((long)NTOK * ED), 256, 0, stream>>>(gs, us, (long)NTOK * ED);
    gemm(1, 0, gs, c.sd, sot, NTOK, DIM, ED, ED, DIM, DIM, 0, 0, 0, 1, 1.f);
    k_moe_combine<<<gblk(ND), 256, 0, stream>>>(ym, wfl, sot, mo);
    k_add<<<gblk(ND), 256, 0, stream>>>(xdst, mo, ND);
  };

  // -------- forward --------
  k_embed<<<NTOK, 256, 0, stream>>>(tokens, embed, h0);
  run_block(h0, h1, bases[0], cs[0]);                 // prelude
  k_copy<<<gblk(ND), 256, 0, stream>>>(ebuf, h1, ND); // e = h
  k_avec<<<gblk(DIM), 256, 0, stream>>>(lti_logA, lti_logdt, Avec);
  hipMemsetAsync(cum, 0, sizeof(float) * NTOK, stream);
  hipMemsetAsync(oac, 0, sizeof(float) * ND, stream);

  for (int t = 0; t < T_LOOP; ++t) {
    k_loop_emb<<<gblk(ND), 256, 0, stream>>>(h1, ht, t);
    run_block(ht, tb, bases[1], cs[1]);
    // LoRA: (ht @ down * scale[t]) @ B
    gemm(1, 0, ht, ldown_bf, t16, NTOK, LORA_R, DIM, DIM, LORA_R, LORA_R, 0, 0, 0, 1, 1.f);
    k_scale_cols<<<gblk((long)NTOK * LORA_R), 256, 0, stream>>>(t16, lora_scale + t * LORA_R, (long)NTOK * LORA_R);
    gemm(1, 0, t16, lB_bf, lor, NTOK, DIM, LORA_R, LORA_R, DIM, DIM, 0, 0, 0, 1, 1.f);
    k_add<<<gblk(ND), 256, 0, stream>>>(tb, lor, ND);
    // h = A*h + B*e + tb
    k_recurrence<<<gblk(ND), 256, 0, stream>>>(h1, Avec, lti_B, ebuf, tb);
    // halting
    k_gemv<<<NTOK, 256, 0, stream>>>(h1, halt_w, halt_b, hw, DIM, 1);
    k_halt<<<gblk(NTOK), 256, 0, stream>>>(hw, cum, wv, t);
    k_axpy_rows<<<gblk(ND), 256, 0, stream>>>(oac, wv, h1);
  }

  run_block(oac, h0, bases[2], cs[2]);                // coda
  k_rmsnorm<<<NTOK, 256, 0, stream>>>(h0, norm_f, xn, DIM, DIM, DIM);
  // logits = h @ embed^T  (embed is V x DIM row-major -> transposed B)
  gemm(1, 1, xn, embed_bf, (float*)d_out, NTOK, V_SZ, DIM, DIM, DIM, V_SZ, 0, 0, 0, 1, 1.f);
}
